// GraphEncoderLayer_14628658610668
// MI455X (gfx1250) — matrix operand order repaired and checked by reading, not compiled, not run on hardware
//
#include <hip/hip_runtime.h>

// ---------------------------------------------------------------------------
// GraphEncoderLayer for MI455X (gfx1250, wave32, WMMA)
// N=50000 nodes, D=128, H=8 heads (HD=16), F=512, E=800000 (+N self loops)
// GEMMs run on v_wmma_f32_16x16x32_bf16 (fp32 inputs downcast to bf16,
// f32 accumulate). Each wave computes a 16x64 output tile (4 accumulators),
// reusing one A fragment across 4 WMMAs per K-step.
// Edge softmax/aggregation via L2-resident gathers + float atomics.
// ---------------------------------------------------------------------------

typedef __attribute__((ext_vector_type(8)))  float  v8f;
typedef __attribute__((ext_vector_type(16))) __bf16 v16bf;
typedef __attribute__((ext_vector_type(8)))  __bf16 v8bf;

#define DCH 128
#define NH  8
#define HD  16
#define NT  4           // column tiles (16 cols each) per wave -> 16x64 tile

__device__ __forceinline__ unsigned short f32_to_bf16_rne(float f) {
    unsigned int u = __float_as_uint(f);
    u += 0x7FFFu + ((u >> 16) & 1u);   // round to nearest even
    return (unsigned short)(u >> 16);
}

__device__ __forceinline__ void atomicMaxFloat(float* addr, float val) {
    if (!signbit(val)) atomicMax((int*)addr, __float_as_int(val));
    else               atomicMin((unsigned int*)addr, __float_as_uint(val));
}

// ------------------------------- casts -------------------------------------
__global__ void cast_f32_to_bf16(const float* __restrict__ in,
                                 unsigned short* __restrict__ out, long n) {
    long i = (long)blockIdx.x * blockDim.x + threadIdx.x;
    if (i < n) out[i] = f32_to_bf16_rne(in[i]);
}

// ----------------------- bf16 WMMA GEMM: Y = X @ W^T -----------------------
// X: MxK bf16 row-major, W: NoutxK bf16 row-major (Y[i,j]=dot(X[i,:],W[j,:])).
// Block = 128 threads = 4 waves; wave w handles row tile blockIdx.x*4+w.
// Each wave owns NT=4 column tiles (16x64 output), K-loop in steps of 32.
// A-frag (16-bit A 16x32): lane m=lane&15, kbase=(lane>>4)*16,
//   elems j = K kbase+j  (contiguous along X row).
// B-frag (16-bit B 32x16): lane n=lane&15, kbase=(lane>>4)*16,
//   elems j = K kbase+j  (contiguous along W row).
// C/D: col=lane&15, rows (lane>>4)*8 + r for accumulator element r.
__global__ __launch_bounds__(128)
void gemm_bf16_wmma(const unsigned short* __restrict__ A,
                    const unsigned short* __restrict__ W,
                    const float* __restrict__ bias,
                    float scale, int leaky,
                    float* __restrict__ outf,
                    unsigned short* __restrict__ outb,
                    int M, int K, int Nout) {
    const int wave = threadIdx.x >> 5;
    const int lane = threadIdx.x & 31;
    const int half = lane >> 4;
    const int l15  = lane & 15;
    const int tm   = blockIdx.x * 4 + wave;          // 16-row tile index
    if (tm * 16 >= M) return;                        // uniform per wave
    const int colBase = blockIdx.y * (16 * NT);

    const unsigned short* arow = A + (size_t)(tm * 16 + l15) * K + half * 16;
    const unsigned short* brow = W + (size_t)(colBase + l15) * K + half * 16;

    v8f c[NT];
#pragma unroll
    for (int t = 0; t < NT; ++t) c[t] = (v8f){};

    for (int kb = 0; kb < K; kb += 32) {
        v8bf lo = *(const v8bf*)(arow + kb);
        v8bf hi = *(const v8bf*)(arow + kb + 8);
        v16bf a;
#pragma unroll
        for (int i = 0; i < 8; ++i) { a[i] = lo[i]; a[i + 8] = hi[i]; }
#pragma unroll
        for (int t = 0; t < NT; ++t) {
            v16bf b = *(const v16bf*)(brow + (size_t)t * 16 * K + kb);
            c[t] = __builtin_amdgcn_wmma_f32_16x16x32_bf16(
                       /*neg_a=*/false, a, /*neg_b=*/false, b,
                       /*c_mod=*/(short)0, c[t],
                       /*reuse_a=*/false, /*reuse_b=*/false);
        }
    }

#pragma unroll
    for (int t = 0; t < NT; ++t) {
        const int col = colBase + t * 16 + l15;
        const float bv = bias ? bias[col] : 0.0f;
#pragma unroll
        for (int r = 0; r < 8; ++r) {
            int row = tm * 16 + half * 8 + r;
            float val = (c[t][r] + bv) * scale;
            if (leaky) val = val > 0.0f ? val : 0.01f * val;
            size_t o = (size_t)row * Nout + col;
            if (outf) outf[o] = val;
            if (outb) outb[o] = f32_to_bf16_rne(val);
        }
    }
}

// ------------------------------ edge helpers -------------------------------
__device__ __forceinline__ void edge_sd(const long long* __restrict__ ei,
                                        long E0, long e, int& s, int& d) {
    if (e < E0) { s = (int)ei[e]; d = (int)ei[E0 + e]; }
    else        { s = d = (int)(e - E0); }
}

__global__ void init_softmax(float* __restrict__ agg, float* __restrict__ mx,
                             float* __restrict__ den, long nAgg, long nSeg) {
    long i = (long)blockIdx.x * blockDim.x + threadIdx.x;
    if (i < nAgg) agg[i] = 0.0f;
    if (i < nSeg) { mx[i] = -1e30f; den[i] = 0.0f; }
}

// scores[e,h] = dot(q[dst,h,:], k[src,h,:]); segment max by dst
__global__ void edge_scores(const long long* __restrict__ ei, long E0, long E,
                            const float* __restrict__ q,
                            const float* __restrict__ k,
                            float* __restrict__ scores,
                            float* __restrict__ mx) {
    long tid = (long)blockIdx.x * blockDim.x + threadIdx.x;
    if (tid >= E * NH) return;
    long e = tid >> 3; int h = (int)(tid & 7);
    int s, d; edge_sd(ei, E0, e, s, d);
    const float4* qp = (const float4*)(q + (size_t)d * DCH + h * HD);
    const float4* kp = (const float4*)(k + (size_t)s * DCH + h * HD);
    float acc = 0.0f;
#pragma unroll
    for (int i = 0; i < 4; ++i) {
        float4 a = qp[i], b = kp[i];
        acc += a.x * b.x + a.y * b.y + a.z * b.z + a.w * b.w;
    }
    scores[tid] = acc;
    atomicMaxFloat(&mx[(size_t)d * NH + h], acc);
}

// scores <- exp(scores - mx[dst]); den[dst] += exp
__global__ void edge_exp(const long long* __restrict__ ei, long E0, long E,
                         float* __restrict__ scores,
                         const float* __restrict__ mx,
                         float* __restrict__ den) {
    long tid = (long)blockIdx.x * blockDim.x + threadIdx.x;
    if (tid >= E * NH) return;
    long e = tid >> 3; int h = (int)(tid & 7);
    int s, d; edge_sd(ei, E0, e, s, d);
    float ex = __expf(scores[tid] - mx[(size_t)d * NH + h]);
    scores[tid] = ex;
    atomicAdd(&den[(size_t)d * NH + h], ex);
}

// agg[dst,h,:] += (ex/den) * v[src,h,:]
__global__ void edge_aggregate(const long long* __restrict__ ei, long E0, long E,
                               const float* __restrict__ scores,
                               const float* __restrict__ den,
                               const float* __restrict__ v,
                               float* __restrict__ agg) {
    long tid = (long)blockIdx.x * blockDim.x + threadIdx.x;
    if (tid >= E * NH) return;
    long e = tid >> 3; int h = (int)(tid & 7);
    int s, d; edge_sd(ei, E0, e, s, d);
    float a = scores[tid] / (den[(size_t)d * NH + h] + 1e-16f);
    const float* vp = v + (size_t)s * DCH + h * HD;
    float* op = agg + (size_t)d * DCH + h * HD;
#pragma unroll
    for (int i = 0; i < HD; ++i) atomicAdd(&op[i], a * vp[i]);
}

// ------------------------------- batchnorm ---------------------------------
__global__ void zero_stats(float* __restrict__ sums, float* __restrict__ sumsq) {
    int c = threadIdx.x;
    if (c < DCH) { sums[c] = 0.0f; sumsq[c] = 0.0f; }
}

#define BN_ROWS 256
__global__ void bn_reduce(const float* __restrict__ a, const float* __restrict__ b,
                          float* __restrict__ sums, float* __restrict__ sumsq,
                          int nrows) {
    int c  = threadIdx.x;                // 0..127 (feature per lane, coalesced)
    int r0 = blockIdx.x * BN_ROWS;
    int r1 = min(r0 + BN_ROWS, nrows);
    float s = 0.0f, s2 = 0.0f;
    for (int r = r0; r < r1; ++r) {
        float vv = a[(size_t)r * DCH + c] + b[(size_t)r * DCH + c];
        s += vv; s2 += vv * vv;
    }
    atomicAdd(&sums[c], s);
    atomicAdd(&sumsq[c], s2);
}

__global__ void bn_finalize(float* __restrict__ sums, float* __restrict__ sumsq,
                            float invN) {
    int c = threadIdx.x;
    if (c < DCH) {
        float mean = sums[c] * invN;
        float var  = sumsq[c] * invN - mean * mean;
        sums[c]  = mean;                 // reuse as mean
        sumsq[c] = rsqrtf(var + 1e-5f);  // reuse as rstd
    }
}

__global__ void bn_apply(const float* __restrict__ a, const float* __restrict__ b,
                         const float* __restrict__ mean, const float* __restrict__ rstd,
                         const float* __restrict__ g, const float* __restrict__ bt,
                         float* __restrict__ outf, unsigned short* __restrict__ outb,
                         long n) {
    long i = (long)blockIdx.x * blockDim.x + threadIdx.x;
    if (i >= n) return;
    int c = (int)(i & (DCH - 1));
    float y = (a[i] + b[i] - mean[c]) * rstd[c] * g[c] + bt[c];
    if (outf) outf[i] = y;
    if (outb) outb[i] = f32_to_bf16_rne(y);
}

// --------------------------------- launch ----------------------------------
extern "C" void kernel_launch(void* const* d_in, const int* in_sizes, int n_in,
                              void* d_out, int out_size, void* d_ws, size_t ws_size,
                              hipStream_t stream) {
    const float*     src = (const float*)d_in[0];
    const long long* ei  = (const long long*)d_in[1];
    const float* q_w = (const float*)d_in[2];
    const float* q_b = (const float*)d_in[3];
    const float* k_w = (const float*)d_in[4];
    const float* v_w = (const float*)d_in[5];
    const float* o_w = (const float*)d_in[6];
    const float* o_b = (const float*)d_in[7];
    const float* g1  = (const float*)d_in[8];
    const float* b1  = (const float*)d_in[9];
    const float* w1  = (const float*)d_in[10];
    const float* l1b = (const float*)d_in[11];
    const float* w2  = (const float*)d_in[12];
    const float* l2b = (const float*)d_in[13];
    const float* g2  = (const float*)d_in[14];
    const float* b2  = (const float*)d_in[15];

    const int  N  = in_sizes[0] / DCH;          // 50000 (multiple of 16)
    const long E0 = in_sizes[1] / 2;            // 800000
    const long E  = E0 + N;                     // + self loops
    const int  F  = in_sizes[10] / DCH;         // 512

    // ---- workspace bump allocator ----
    size_t off = 0;
    auto alloc = [&](size_t bytes) -> void* {
        void* p = (char*)d_ws + off;
        off += (bytes + 255) & ~(size_t)255;
        return p;
    };
    unsigned short* src_bf = (unsigned short*)alloc((size_t)N * DCH * 2);
    unsigned short* wq_bf  = (unsigned short*)alloc((size_t)DCH * DCH * 2);
    unsigned short* wk_bf  = (unsigned short*)alloc((size_t)DCH * DCH * 2);
    unsigned short* wv_bf  = (unsigned short*)alloc((size_t)DCH * DCH * 2);
    unsigned short* wo_bf  = (unsigned short*)alloc((size_t)DCH * DCH * 2);
    unsigned short* w1_bf  = (unsigned short*)alloc((size_t)F * DCH * 2);
    unsigned short* w2_bf  = (unsigned short*)alloc((size_t)DCH * F * 2);
    float* q    = (float*)alloc((size_t)N * DCH * 4);
    float* k    = (float*)alloc((size_t)N * DCH * 4);
    float* v    = (float*)alloc((size_t)N * DCH * 4);
    float* sc   = (float*)alloc((size_t)E * NH * 4);
    float* mx   = (float*)alloc((size_t)N * NH * 4);
    float* den  = (float*)alloc((size_t)N * NH * 4);
    float* agg  = (float*)alloc((size_t)N * DCH * 4);
    unsigned short* agg_bf = (unsigned short*)alloc((size_t)N * DCH * 2);
    float* aout = (float*)alloc((size_t)N * DCH * 4);
    float* x    = (float*)alloc((size_t)N * DCH * 4);
    unsigned short* x_bf  = (unsigned short*)alloc((size_t)N * DCH * 2);
    unsigned short* h1_bf = (unsigned short*)alloc((size_t)N * F * 2);
    float* ff   = (float*)alloc((size_t)N * DCH * 4);
    float* bns  = (float*)alloc(DCH * 4);
    float* bnq  = (float*)alloc(DCH * 4);
    (void)ws_size; (void)n_in; (void)out_size;

    auto blocks = [](long n, int bs) { return (int)((n + bs - 1) / bs); };

    // ---- fp32 -> bf16 casts ----
    cast_f32_to_bf16<<<blocks((long)N * DCH, 256), 256, 0, stream>>>(src, src_bf, (long)N * DCH);
    cast_f32_to_bf16<<<blocks((long)DCH * DCH, 256), 256, 0, stream>>>(q_w, wq_bf, (long)DCH * DCH);
    cast_f32_to_bf16<<<blocks((long)DCH * DCH, 256), 256, 0, stream>>>(k_w, wk_bf, (long)DCH * DCH);
    cast_f32_to_bf16<<<blocks((long)DCH * DCH, 256), 256, 0, stream>>>(v_w, wv_bf, (long)DCH * DCH);
    cast_f32_to_bf16<<<blocks((long)DCH * DCH, 256), 256, 0, stream>>>(o_w, wo_bf, (long)DCH * DCH);
    cast_f32_to_bf16<<<blocks((long)F * DCH, 256), 256, 0, stream>>>(w1, w1_bf, (long)F * DCH);
    cast_f32_to_bf16<<<blocks((long)DCH * F, 256), 256, 0, stream>>>(w2, w2_bf, (long)DCH * F);

    // ---- Q/K/V projections (WMMA): grid = (row tiles / 4 waves, cols/64) ----
    const int rowBlocks = (N / 16 + 3) / 4;
    dim3 gD(rowBlocks, DCH / 64);
    const float scaling = 0.25f;   // HD^-0.5, HD=16
    gemm_bf16_wmma<<<gD, 128, 0, stream>>>(src_bf, wq_bf, q_b, scaling, 0, q, nullptr, N, DCH, DCH);
    gemm_bf16_wmma<<<gD, 128, 0, stream>>>(src_bf, wk_bf, nullptr, 1.0f, 0, k, nullptr, N, DCH, DCH);
    gemm_bf16_wmma<<<gD, 128, 0, stream>>>(src_bf, wv_bf, nullptr, 1.0f, 0, v, nullptr, N, DCH, DCH);

    // ---- segment softmax attention over edges ----
    init_softmax<<<blocks((long)N * DCH, 256), 256, 0, stream>>>(agg, mx, den, (long)N * DCH, (long)N * NH);
    long EH = E * NH;
    edge_scores<<<blocks(EH, 256), 256, 0, stream>>>(ei, E0, E, q, k, sc, mx);
    edge_exp   <<<blocks(EH, 256), 256, 0, stream>>>(ei, E0, E, sc, mx, den);
    edge_aggregate<<<blocks(EH, 256), 256, 0, stream>>>(ei, E0, E, sc, den, v, agg);

    // ---- output projection + BN1 ----
    cast_f32_to_bf16<<<blocks((long)N * DCH, 256), 256, 0, stream>>>(agg, agg_bf, (long)N * DCH);
    gemm_bf16_wmma<<<gD, 128, 0, stream>>>(agg_bf, wo_bf, o_b, 1.0f, 0, aout, nullptr, N, DCH, DCH);

    const float invN = 1.0f / (float)N;
    zero_stats<<<1, DCH, 0, stream>>>(bns, bnq);
    bn_reduce<<<blocks(N, BN_ROWS), DCH, 0, stream>>>(src, aout, bns, bnq, N);
    bn_finalize<<<1, DCH, 0, stream>>>(bns, bnq, invN);
    bn_apply<<<blocks((long)N * DCH, 256), 256, 0, stream>>>(src, aout, bns, bnq, g1, b1,
                                                             x, x_bf, (long)N * DCH);

    // ---- FFN (WMMA) + BN2 -> d_out ----
    dim3 gF(rowBlocks, F / 64);
    gemm_bf16_wmma<<<gF, 128, 0, stream>>>(x_bf, w1_bf, l1b, 1.0f, /*leaky=*/1, nullptr, h1_bf, N, DCH, F);
    gemm_bf16_wmma<<<gD, 128, 0, stream>>>(h1_bf, w2_bf, l2b, 1.0f, 0, ff, nullptr, N, F, DCH);

    zero_stats<<<1, DCH, 0, stream>>>(bns, bnq);
    bn_reduce<<<blocks(N, BN_ROWS), DCH, 0, stream>>>(x, ff, bns, bnq, N);
    bn_finalize<<<1, DCH, 0, stream>>>(bns, bnq, invN);
    bn_apply<<<blocks((long)N * DCH, 256), 256, 0, stream>>>(x, ff, bns, bnq, g2, b2,
                                                             (float*)d_out, nullptr, (long)N * DCH);
}
